// AttentiveTransformer_67662914781376
// MI455X (gfx1250) — compile-verified, matrix operand order checked by reading, not run on hardware
//
#include <hip/hip_runtime.h>
#include <hip/hip_bf16.h>

#define BB   262144
#define NA   128
#define DD   256
#define EPSV 1e-5f

typedef __attribute__((ext_vector_type(16))) __bf16 v16bf;
typedef __attribute__((ext_vector_type(8)))  float  v8f;
typedef __attribute__((ext_vector_type(4)))  unsigned int u32x4;
typedef __attribute__((ext_vector_type(8)))  int         i32x8;

// ---------------------------------------------------------------------------
// Kernel 1: split W (fp32, row-major [D][NA]) into bf16 hi/lo; zero stat accums.
// ---------------------------------------------------------------------------
__global__ void k_prep(const float* __restrict__ W, __bf16* __restrict__ Whi,
                       __bf16* __restrict__ Wlo, float* __restrict__ gsum,
                       float* __restrict__ gsq) {
    int i = blockIdx.x * 256 + threadIdx.x;        // 128 blocks * 256 = D*NA
    float w = W[i];
    __bf16 h = (__bf16)w;
    Whi[i] = h;
    Wlo[i] = (__bf16)(w - (float)h);
    if (blockIdx.x == 0) { gsum[threadIdx.x] = 0.0f; gsq[threadIdx.x] = 0.0f; }
}

// ---------------------------------------------------------------------------
// TDM: DMA the block's x tile (32 rows x 128 fp32, row stride 128) into LDS.
// D# built in SGPRs per CDNA5 ISA; 2D tile -> two SGPR
// groups only. Issued by one wave; completion via TENSORcnt.
// ---------------------------------------------------------------------------
__device__ __forceinline__ void tdm_load_x(unsigned lds_off, const float* gptr) {
    unsigned long long ga = (unsigned long long)(uintptr_t)gptr;
    u32x4 g0;
    g0.x = 1u;                                          // count=1, user mode
    g0.y = lds_off;                                     // lds_addr (bytes)
    g0.z = (unsigned)(ga & 0xFFFFFFFFull);              // global_addr[31:0]
    g0.w = (unsigned)((ga >> 32) & 0x01FFFFFFull)       // global_addr[56:32]
           | (2u << 30);                                // type=2 ("image")
    i32x8 g1;
    g1[0] = (int)(2u << 16);      // workgroup_mask=0, data_size=2 (4 bytes)
    g1[1] = (int)(128u << 16);    // tensor_dim0[15:0]=128 at bits [63:48]
    g1[2] = (int)(32u << 16);     // tensor_dim0[31:16]=0 | tensor_dim1[15:0]=32
    g1[3] = (int)(128u << 16);    // tensor_dim1[31:16]=0 | tile_dim0=128
    g1[4] = 32;                   // tile_dim1=32, tile_dim2=0
    g1[5] = 128;                  // tensor_dim0_stride[31:0] = 128 elements
    g1[6] = 0;                    // stride hi / tensor_dim1_stride lo
    g1[7] = 0;
    asm volatile("tensor_load_to_lds %0, %1" :: "s"(g0), "s"(g1) : "memory");
}

// ---------------------------------------------------------------------------
// Shared WMMA GEMM core: one wave computes a 16(M) x 64(N) tile of z = x*W^T
// using split-bf16 (hi/lo). A-fragments come from the LDS-staged x tile.
// A layout (16-bit, 16x32): lane L -> M = L%16, khalf = L/16; two contiguous
// 8-element K-chunks at K0 = k + 8*khalf and K0+16. B mirrors with N = L%16.
// ---------------------------------------------------------------------------
__device__ __forceinline__ void wave_gemm(const float* __restrict__ xs,   // LDS
                                          const __bf16* __restrict__ Whi,
                                          const __bf16* __restrict__ Wlo,
                                          int m0l, int nbase, int lane,
                                          v8f acc[4]) {
    const int mr = lane & 15;
    const int kh = lane >> 4;
    const float* xrow = xs + (m0l + mr) * NA;

    v8f zero = {0.f, 0.f, 0.f, 0.f, 0.f, 0.f, 0.f, 0.f};
#pragma unroll
    for (int t = 0; t < 4; ++t) acc[t] = zero;

#pragma unroll
    for (int k = 0; k < NA; k += 32) {
        const int K0 = k + 8 * kh;
        const int K1 = K0 + 16;

        // ---- A tile: 16 fp32 (from LDS) -> bf16 hi/lo
        float4 a0 = *(const float4*)(xrow + K0);
        float4 a1 = *(const float4*)(xrow + K0 + 4);
        float4 a2 = *(const float4*)(xrow + K1);
        float4 a3 = *(const float4*)(xrow + K1 + 4);
        float fa[16] = {a0.x, a0.y, a0.z, a0.w, a1.x, a1.y, a1.z, a1.w,
                        a2.x, a2.y, a2.z, a2.w, a3.x, a3.y, a3.z, a3.w};
        v16bf ahi, alo;
#pragma unroll
        for (int i = 0; i < 16; ++i) {
            __bf16 h = (__bf16)fa[i];
            ahi[i] = h;
            alo[i] = (__bf16)(fa[i] - (float)h);
        }

#pragma unroll
        for (int t = 0; t < 4; ++t) {
            const int N = nbase + 16 * t + mr;
            const __bf16* wh = Whi + (size_t)N * NA;
            const __bf16* wl = Wlo + (size_t)N * NA;
            union U { uint4 u[2]; v16bf v; };
            U bh, bl;
            bh.u[0] = *(const uint4*)(wh + K0);
            bh.u[1] = *(const uint4*)(wh + K1);
            bl.u[0] = *(const uint4*)(wl + K0);
            bl.u[1] = *(const uint4*)(wl + K1);

            acc[t] = __builtin_amdgcn_wmma_f32_16x16x32_bf16(
                false, ahi, false, bh.v, (short)0, acc[t], false, false);
            acc[t] = __builtin_amdgcn_wmma_f32_16x16x32_bf16(
                false, ahi, false, bl.v, (short)0, acc[t], false, false);
            acc[t] = __builtin_amdgcn_wmma_f32_16x16x32_bf16(
                false, alo, false, bh.v, (short)0, acc[t], false, false);
        }
    }
}

// ---------------------------------------------------------------------------
// Kernel 2: GEMM + per-column batch statistics (sum, sum of squares).
// Block = 256 threads = 8 waves; tile = 32 batch rows x 256 columns.
// z tiles never touch HBM.
// ---------------------------------------------------------------------------
__global__ void k_gemm_stats(const float* __restrict__ x,
                             const __bf16* __restrict__ Whi,
                             const __bf16* __restrict__ Wlo,
                             float* __restrict__ gsum, float* __restrict__ gsq) {
    __shared__ float xs[32 * NA];                  // 16 KB TDM-staged x tile
    __shared__ float lsum[DD];
    __shared__ float lsq[DD];
    const int tid  = threadIdx.x;
    const int wave = tid >> 5;
    const int lane = tid & 31;
    const int b0   = blockIdx.x * 32;

    if (tid < 32) {
        tdm_load_x((unsigned)(uintptr_t)xs, x + (size_t)b0 * NA);
        __builtin_amdgcn_s_wait_tensorcnt(0);
    }
    lsum[tid] = 0.0f;
    lsq[tid]  = 0.0f;
    __syncthreads();

    const int m0l   = (wave >> 2) * 16;
    const int nbase = (wave & 3) * 64;
    v8f acc[4];
    wave_gemm(xs, Whi, Wlo, m0l, nbase, lane, acc);

#pragma unroll
    for (int t = 0; t < 4; ++t) {
        float s = 0.0f, q = 0.0f;
#pragma unroll
        for (int v = 0; v < 8; ++v) { float z = acc[t][v]; s += z; q += z * z; }
        s += __shfl_xor(s, 16);   // fold the two M-halves of the 16x16 tile
        q += __shfl_xor(q, 16);
        if (lane < 16) {
            int n = nbase + 16 * t + lane;
            atomicAdd(&lsum[n], s);   // ds_add_f32
            atomicAdd(&lsq[n], q);
        }
    }
    __syncthreads();
    atomicAdd(&gsum[tid], lsum[tid]); // one global f32 atomic per column/block
    atomicAdd(&gsq[tid],  lsq[tid]);
}

// ---------------------------------------------------------------------------
// Kernel 3: fold BN into per-column scale/shift. (fc bias cancels inside BN.)
// ---------------------------------------------------------------------------
__global__ void k_stats_final(const float* __restrict__ gsum,
                              const float* __restrict__ gsq,
                              const float* __restrict__ gamma,
                              const float* __restrict__ beta,
                              float* __restrict__ scale,
                              float* __restrict__ shift) {
    int d = threadIdx.x;
    const float inv = 1.0f / (float)BB;
    float mean = gsum[d] * inv;
    float var  = gsq[d] * inv - mean * mean;
    float s    = gamma[d] * rsqrtf(var + EPSV);
    scale[d] = s;
    shift[d] = beta[d] - mean * s;
}

// ---------------------------------------------------------------------------
// Kernel 4: GEMM (recompute; cheaper than a 512 MB HBM round-trip for z) + BN
// into LDS + row-wise sparsemax (bisection + exact top-k refinement).
// ---------------------------------------------------------------------------
__global__ void k_gemm_sparsemax(const float* __restrict__ x,
                                 const __bf16* __restrict__ Whi,
                                 const __bf16* __restrict__ Wlo,
                                 const float* __restrict__ scale,
                                 const float* __restrict__ shift,
                                 const float* __restrict__ prior,
                                 float* __restrict__ out_sm,
                                 float* __restrict__ out_np) {
    __shared__ float xs[32 * NA];                  // 16 KB TDM-staged x tile
    __shared__ float zn[32 * 264];                 // 33 KB, padded row stride
    const int tid  = threadIdx.x;
    const int wave = tid >> 5;
    const int lane = tid & 31;
    const int mrow  = wave >> 2;                   // 0..1
    const int nbase = (wave & 3) * 64;
    const int b0 = blockIdx.x * 32;

    if (tid < 32) {
        tdm_load_x((unsigned)(uintptr_t)xs, x + (size_t)b0 * NA);
        __builtin_amdgcn_s_wait_tensorcnt(0);
    }
    __syncthreads();

    v8f acc[4];
    wave_gemm(xs, Whi, Wlo, mrow * 16, nbase, lane, acc);

    // scatter BN-normalized z into LDS (C/D layout: lane -> N=L%16, M=8*(L/16)+v)
    const int mr = lane & 15, mh = lane >> 4;
#pragma unroll
    for (int t = 0; t < 4; ++t) {
        const int n = nbase + 16 * t + mr;
        const float sc = scale[n];
        const float sh = shift[n];
#pragma unroll
        for (int v = 0; v < 8; ++v) {
            const int M = mrow * 16 + mh * 8 + v;
            zn[M * 264 + n] = acc[t][v] * sc + sh;
        }
    }
    __syncthreads();

    // ---- sparsemax: 8 threads per row, 32 values per thread (kept in VGPRs)
    const int r = tid >> 3;                        // 0..31 row in tile
    const int c = (tid & 7) * 32;                  // column chunk base
    const float* prow = prior + (size_t)(b0 + r) * DD + c;

    float y[32], p[32];
#pragma unroll
    for (int i = 0; i < 32; i += 4) {
        float4 pv = *(const float4*)(prow + i);
        p[i] = pv.x; p[i + 1] = pv.y; p[i + 2] = pv.z; p[i + 3] = pv.w;
    }
#pragma unroll
    for (int i = 0; i < 32; ++i) y[i] = zn[r * 264 + c + i] * p[i];

    // row max over 256 values (thread-local + 8-lane shuffle tree)
    float mx = -3.4e38f;
#pragma unroll
    for (int i = 0; i < 32; ++i) mx = fmaxf(mx, y[i]);
    mx = fmaxf(mx, __shfl_xor(mx, 1));
    mx = fmaxf(mx, __shfl_xor(mx, 2));
    mx = fmaxf(mx, __shfl_xor(mx, 4));

    // bisection on f(tau) = sum(relu(y - tau)) - 1, root in [mx-1, mx]
    float lo = mx - 1.0f, hi = mx;
#pragma unroll 1
    for (int it = 0; it < 24; ++it) {
        float tau = 0.5f * (lo + hi);
        float s = 0.0f;
#pragma unroll
        for (int i = 0; i < 32; ++i) s += fmaxf(y[i] - tau, 0.0f);
        s += __shfl_xor(s, 1);
        s += __shfl_xor(s, 2);
        s += __shfl_xor(s, 4);
        if (s > 1.0f) lo = tau; else hi = tau;
    }

    // exact refinement: tau = (sum_{y>tau0} y - 1) / k
    const float tau0 = 0.5f * (lo + hi);
    float ssum = 0.0f, cnt = 0.0f;
#pragma unroll
    for (int i = 0; i < 32; ++i)
        if (y[i] > tau0) { ssum += y[i]; cnt += 1.0f; }
    ssum += __shfl_xor(ssum, 1); cnt += __shfl_xor(cnt, 1);
    ssum += __shfl_xor(ssum, 2); cnt += __shfl_xor(cnt, 2);
    ssum += __shfl_xor(ssum, 4); cnt += __shfl_xor(cnt, 4);
    const float tau = (ssum - 1.0f) / fmaxf(cnt, 1.0f);

    float* so = out_sm + (size_t)(b0 + r) * DD + c;
    float* po = out_np + (size_t)(b0 + r) * DD + c;
#pragma unroll
    for (int i = 0; i < 32; i += 4) {
        float4 sv, nv;
        sv.x = fmaxf(y[i]     - tau, 0.0f); nv.x = p[i]     * sv.x;
        sv.y = fmaxf(y[i + 1] - tau, 0.0f); nv.y = p[i + 1] * sv.y;
        sv.z = fmaxf(y[i + 2] - tau, 0.0f); nv.z = p[i + 2] * sv.z;
        sv.w = fmaxf(y[i + 3] - tau, 0.0f); nv.w = p[i + 3] * sv.w;
        *(float4*)(so + i) = sv;
        *(float4*)(po + i) = nv;
    }
}

// ---------------------------------------------------------------------------
extern "C" void kernel_launch(void* const* d_in, const int* in_sizes, int n_in,
                              void* d_out, int out_size, void* d_ws, size_t ws_size,
                              hipStream_t stream) {
    const float* x     = (const float*)d_in[0];
    const float* prior = (const float*)d_in[1];
    const float* W     = (const float*)d_in[2];
    const float* bias  = (const float*)d_in[3]; // cancels exactly inside BN
    const float* gamma = (const float*)d_in[4];
    const float* beta  = (const float*)d_in[5];
    (void)bias; (void)in_sizes; (void)n_in; (void)out_size; (void)ws_size;

    char* ws = (char*)d_ws;
    __bf16* Whi  = (__bf16*)ws;                       //  64 KB
    __bf16* Wlo  = (__bf16*)(ws + 65536);             //  64 KB
    float*  gsum = (float*)(ws + 131072);             //   1 KB
    float*  gsq  = (float*)(ws + 131072 + 1024);      //   1 KB
    float*  scl  = (float*)(ws + 131072 + 2048);      //   1 KB
    float*  shf  = (float*)(ws + 131072 + 3072);      //   1 KB

    float* out_sm = (float*)d_out;
    float* out_np = out_sm + (size_t)BB * DD;

    k_prep<<<(DD * NA) / 256, 256, 0, stream>>>(W, Whi, Wlo, gsum, gsq);
    k_gemm_stats<<<BB / 32, 256, 0, stream>>>(x, Whi, Wlo, gsum, gsq);
    k_stats_final<<<1, DD, 0, stream>>>(gsum, gsq, gamma, beta, scl, shf);
    k_gemm_sparsemax<<<BB / 32, 256, 0, stream>>>(x, Whi, Wlo, scl, shf, prior,
                                                  out_sm, out_np);
}
